// MultiHeadAttention_13142599926360
// MI455X (gfx1250) — compile-verified, matrix operand order checked by reading, not run on hardware
//
#include <hip/hip_runtime.h>
#include <hip/hip_bf16.h>
#include <math.h>

#define BB 8
#define SS 1024
#define DM 1024
#define NH 16
#define DK 64
#define DV 64

typedef __attribute__((ext_vector_type(16))) __bf16 v16bf;
typedef __attribute__((ext_vector_type(8)))  __bf16 v8bf;
typedef __attribute__((ext_vector_type(8)))  float  v8f;

// Load a 16-element bf16 fragment per lane: elements 0..7 = K kb..kb+7,
// elements 8..15 = K kb+16..kb+23 (ISA 16-bit A/B layout, wave32).
__device__ __forceinline__ v16bf load_frag_g(const __bf16* p) {
  v8bf lo = *(const v8bf*)(p);
  v8bf hi = *(const v8bf*)(p + 16);
  v16bf r;
#pragma unroll
  for (int i = 0; i < 8; ++i) { r[i] = lo[i]; r[i + 8] = hi[i]; }
  return r;
}

__device__ __forceinline__ v8f wmma_bf16(v16bf a, v16bf b, v8f c) {
  return __builtin_amdgcn_wmma_f32_16x16x32_bf16(false, a, false, b, (short)0, c,
                                                 false, false);
}

// ---------------- conversion kernels ----------------
__global__ void cvt_f32_bf16(const float* __restrict__ in, __bf16* __restrict__ out, int n) {
  int i = blockIdx.x * blockDim.x + threadIdx.x;
  int stride = gridDim.x * blockDim.x;
  for (; i < n; i += stride) out[i] = (__bf16)in[i];
}

// w: [NH, DM, DK] f32  ->  out: [NH, DK, DM] bf16 (K-contiguous B fragments)
__global__ void wtrans_bf16(const float* __restrict__ in, __bf16* __restrict__ out) {
  int i = blockIdx.x * blockDim.x + threadIdx.x;
  const int n = NH * DK * DM;
  int stride = gridDim.x * blockDim.x;
  for (; i < n; i += stride) {
    int k = i % DM;
    int t = i / DM;
    int d = t % DK;
    int h = t / DK;
    out[i] = (__bf16)in[((size_t)h * DM + k) * DK + d];
  }
}

// ---------------- per-head QKV projection ----------------
// One wave computes a 16x64 output tile: qh/kh [b,h,s,dk], v stored transposed [b,h,dv,s].
__global__ void __launch_bounds__(32) proj_kernel(
    const __bf16* __restrict__ qbf, const __bf16* __restrict__ kbf,
    const __bf16* __restrict__ vbf, const __bf16* __restrict__ wqT,
    const __bf16* __restrict__ wkT, const __bf16* __restrict__ wvT,
    __bf16* __restrict__ QH, __bf16* __restrict__ KH, __bf16* __restrict__ VT) {
  int wg = blockIdx.x;
  int stile = wg & 63;
  int h = (wg >> 6) & 15;
  int b = (wg >> 10) & 7;
  int mat = wg >> 13;  // 0=q 1=k 2=v
  int lane = threadIdx.x & 31;
  int l16 = lane & 15, half = lane >> 4, kb = half * 8;
  const __bf16* X = (mat == 0) ? qbf : ((mat == 1) ? kbf : vbf);
  const __bf16* W = (mat == 0) ? wqT : ((mat == 1) ? wkT : wvT);
  int m0 = stile * 16;
  const __bf16* Arow = X + ((size_t)b * SS + m0 + l16) * DM;
  const __bf16* Wb = W + (size_t)h * DK * DM;
  v8f c[4] = {{}, {}, {}, {}};
  for (int kk = 0; kk < DM; kk += 32) {
    __builtin_prefetch(Arow + kk + 128, 0, 1);
    v16bf a = load_frag_g(Arow + kk + kb);
#pragma unroll
    for (int t = 0; t < 4; ++t) {
      v16bf bf = load_frag_g(Wb + (size_t)(t * 16 + l16) * DM + kk + kb);
      c[t] = wmma_bf16(a, bf, c[t]);
    }
  }
  float scale = (mat == 0) ? 0.125f : 1.0f;  // fold 1/sqrt(d_k) into Q
  size_t head = (size_t)b * NH + h;
#pragma unroll
  for (int t = 0; t < 4; ++t) {
#pragma unroll
    for (int r = 0; r < 8; ++r) {
      int row = m0 + r + half * 8;
      int col = t * 16 + l16;
      float val = c[t][r] * scale;
      if (mat == 2) {
        VT[(head * DV + col) * SS + row] = (__bf16)val;
      } else {
        __bf16* O = (mat == 0) ? QH : KH;
        O[(head * SS + row) * DK + col] = (__bf16)val;
      }
    }
  }
}

// ---------------- attention: scores + softmax + context ----------------
// 1 workgroup (4 waves) per (b, h, 16-query tile). Full score rows in LDS (64KB).
__global__ void __launch_bounds__(128) attn_kernel(
    const __bf16* __restrict__ QH, const __bf16* __restrict__ KH,
    const __bf16* __restrict__ VT, const unsigned char* __restrict__ mask,
    float* __restrict__ attn_out, __bf16* __restrict__ concat) {
  __shared__ float sc[16 * 1024];
  __shared__ float rmax[16];
  __shared__ float rsum[16];
  int wg = blockIdx.x;
  int qt = wg & 63;
  int h = (wg >> 6) & 15;
  int b = wg >> 10;
  int tid = threadIdx.x;
  int wid = tid >> 5;
  int lane = tid & 31, l16 = lane & 15, half = lane >> 4, kb = half * 8;
  int q0 = qt * 16;
  size_t head = (size_t)b * NH + h;
  const __bf16* Q = QH + head * SS * DK;
  const __bf16* K = KH + head * SS * DK;
  const __bf16* V = VT + head * DV * SS;

  // Q fragments for this tile (d_k = 64 -> two K windows)
  v16bf aq0 = load_frag_g(Q + (size_t)(q0 + l16) * DK + 0 + kb);
  v16bf aq1 = load_frag_g(Q + (size_t)(q0 + l16) * DK + 32 + kb);

  const unsigned char* mbase = mask + (size_t)b * SS * SS;

  // scores: each wave handles every 4th 16-key tile
  for (int kt = wid; kt < 64; kt += 4) {
    int n0 = kt * 16;
    v16bf b0 = load_frag_g(K + (size_t)(n0 + l16) * DK + 0 + kb);
    v16bf b1 = load_frag_g(K + (size_t)(n0 + l16) * DK + 32 + kb);
    v8f c = {};
    c = wmma_bf16(aq0, b0, c);
    c = wmma_bf16(aq1, b1, c);
    int col = n0 + l16;
#pragma unroll
    for (int r = 0; r < 8; ++r) {
      int row = r + half * 8;
      float v = c[r];
      if (mbase[(size_t)(q0 + row) * SS + col]) v = -INFINITY;
      sc[row * 1024 + col] = v;
    }
  }
  __syncthreads();
  if (tid < 16) {
    float m = -INFINITY;
    for (int cx = 0; cx < 1024; ++cx) m = fmaxf(m, sc[tid * 1024 + cx]);
    rmax[tid] = m;
    rsum[tid] = 0.f;
  }
  __syncthreads();
  for (int idx = tid; idx < 16 * 1024; idx += 128) {
    int row = idx >> 10;
    float e = __expf(sc[idx] - rmax[row]);
    sc[idx] = e;  // keep unnormalized exp in LDS
    atomicAdd(&rsum[row], e);  // ds_add_f32
  }
  __syncthreads();
  // normalized attention probabilities -> d_out ([h*B+b, q, k])
  float* aout = attn_out + (((size_t)h * BB + b) * SS + q0) * SS;
  for (int idx = tid; idx < 16 * 1024; idx += 128) {
    int row = idx >> 10;
    aout[(size_t)row * SS + (idx & 1023)] = sc[idx] * (1.0f / rsum[row]);
  }
  // ctx = E x V : each wave computes one 16-wide d_v column tile over all K
  v8f cc = {};
  int n0 = wid * 16;
  const __bf16* Vr = V + (size_t)(n0 + l16) * SS;
  for (int kk = 0; kk < SS; kk += 32) {
    v16bf a;
#pragma unroll
    for (int i = 0; i < 8; ++i) {
      a[i]     = (__bf16)sc[l16 * 1024 + kk + kb + i];
      a[i + 8] = (__bf16)sc[l16 * 1024 + kk + kb + 16 + i];
    }
    v16bf bv = load_frag_g(Vr + kk + kb);
    cc = wmma_bf16(a, bv, cc);
  }
#pragma unroll
  for (int r = 0; r < 8; ++r) {
    int row = r + half * 8;
    float val = cc[r] * (1.0f / rsum[row]);
    concat[((size_t)b * SS + q0 + row) * DM + h * DV + n0 + l16] = (__bf16)val;
  }
}

// ---------------- output projection + bias + residual ----------------
// 4 waves x 4 tiles = 16 rows x 256 cols per workgroup.
__global__ void __launch_bounds__(128) outproj_kernel(
    const __bf16* __restrict__ concat, const __bf16* __restrict__ pw,
    const float* __restrict__ proj_b, const float* __restrict__ qin,
    float* __restrict__ z) {
  int wg = blockIdx.x;
  int nt = wg & 3;
  int st = (wg >> 2) & 63;
  int b = wg >> 8;
  int tid = threadIdx.x;
  int wid = tid >> 5;
  int lane = tid & 31, l16 = lane & 15, half = lane >> 4, kb = half * 8;
  int m0 = st * 16;
  int colbase = nt * 256 + wid * 64;
  const __bf16* Arow = concat + ((size_t)b * SS + m0 + l16) * DM;
  v8f c[4] = {{}, {}, {}, {}};
  for (int kk = 0; kk < DM; kk += 32) {
    __builtin_prefetch(Arow + kk + 128, 0, 1);
    v16bf a = load_frag_g(Arow + kk + kb);
#pragma unroll
    for (int t = 0; t < 4; ++t) {
      v16bf bf = load_frag_g(pw + (size_t)(colbase + t * 16 + l16) * DM + kk + kb);
      c[t] = wmma_bf16(a, bf, c[t]);
    }
  }
#pragma unroll
  for (int t = 0; t < 4; ++t) {
    int col = colbase + t * 16 + l16;
    float bias = proj_b[col];
#pragma unroll
    for (int r = 0; r < 8; ++r) {
      int row = m0 + r + half * 8;
      size_t off = ((size_t)b * SS + row) * DM + col;
      z[off] = c[t][r] + bias + qin[off];
    }
  }
}

// ---------------- LayerNorm (unbiased var, eps on std) ----------------
__global__ void __launch_bounds__(256) ln_kernel(
    const float* __restrict__ z, const float* __restrict__ gamma,
    const float* __restrict__ beta, float* __restrict__ out) {
  __shared__ float red[256];
  int row = blockIdx.x;
  const float* zr = z + (size_t)row * DM;
  int tid = threadIdx.x;
  float s = 0.f;
  for (int i = tid; i < DM; i += 256) s += zr[i];
  red[tid] = s;
  __syncthreads();
  for (int o = 128; o > 0; o >>= 1) {
    if (tid < o) red[tid] += red[tid + o];
    __syncthreads();
  }
  float mean = red[0] * (1.0f / DM);
  __syncthreads();
  float v = 0.f;
  for (int i = tid; i < DM; i += 256) {
    float d = zr[i] - mean;
    v += d * d;
  }
  red[tid] = v;
  __syncthreads();
  for (int o = 128; o > 0; o >>= 1) {
    if (tid < o) red[tid] += red[tid + o];
    __syncthreads();
  }
  float stdv = sqrtf(red[0] * (1.0f / (DM - 1))) + 1e-3f;
  for (int i = tid; i < DM; i += 256)
    out[(size_t)row * DM + i] = gamma[i] * (zr[i] - mean) / stdv + beta[i];
}

extern "C" void kernel_launch(void* const* d_in, const int* in_sizes, int n_in,
                              void* d_out, int out_size, void* d_ws, size_t ws_size,
                              hipStream_t stream) {
  const float* q = (const float*)d_in[0];
  const float* k = (const float*)d_in[1];
  const float* v = (const float*)d_in[2];
  const unsigned char* mask = (const unsigned char*)d_in[3];
  const float* w_q = (const float*)d_in[4];
  const float* w_k = (const float*)d_in[5];
  const float* w_v = (const float*)d_in[6];
  const float* proj_w = (const float*)d_in[7];
  const float* proj_b = (const float*)d_in[8];
  const float* gamma = (const float*)d_in[9];
  const float* beta = (const float*)d_in[10];
  (void)in_sizes; (void)n_in; (void)out_size; (void)ws_size;

  char* ws = (char*)d_ws;
  size_t off = 0;
  auto alloc = [&](size_t bytes) {
    size_t o = off;
    off += (bytes + 255) & ~(size_t)255;
    return o;
  };
  const size_t NXD = (size_t)BB * SS * DM;       // 8M elements
  const size_t NW = (size_t)NH * DM * DK;        // 1M elements
  const size_t NHD = (size_t)BB * NH * SS * DK;  // 8M elements

  __bf16* qbf = (__bf16*)(ws + alloc(NXD * 2));
  __bf16* kbf = (__bf16*)(ws + alloc(NXD * 2));
  __bf16* vbf = (__bf16*)(ws + alloc(NXD * 2));
  __bf16* wqT = (__bf16*)(ws + alloc(NW * 2));
  __bf16* wkT = (__bf16*)(ws + alloc(NW * 2));
  __bf16* wvT = (__bf16*)(ws + alloc(NW * 2));
  __bf16* pwbf = (__bf16*)(ws + alloc((size_t)DM * DM * 2));
  __bf16* QH = (__bf16*)(ws + alloc(NHD * 2));
  __bf16* KH = (__bf16*)(ws + alloc(NHD * 2));
  __bf16* VT = (__bf16*)(ws + alloc(NHD * 2));
  __bf16* concat = (__bf16*)(ws + alloc(NXD * 2));
  float* z = (float*)(ws + alloc(NXD * 4));

  float* ln_out = (float*)d_out;
  float* attn_out = (float*)d_out + NXD;  // [NH*BB, SS, SS] follows LN output

  cvt_f32_bf16<<<4096, 256, 0, stream>>>(q, qbf, (int)NXD);
  cvt_f32_bf16<<<4096, 256, 0, stream>>>(k, kbf, (int)NXD);
  cvt_f32_bf16<<<4096, 256, 0, stream>>>(v, vbf, (int)NXD);
  cvt_f32_bf16<<<1024, 256, 0, stream>>>(proj_w, pwbf, DM * DM);
  wtrans_bf16<<<1024, 256, 0, stream>>>(w_q, wqT);
  wtrans_bf16<<<1024, 256, 0, stream>>>(w_k, wkT);
  wtrans_bf16<<<1024, 256, 0, stream>>>(w_v, wvT);

  proj_kernel<<<3 * BB * NH * (SS / 16), 32, 0, stream>>>(qbf, kbf, vbf, wqT, wkT,
                                                          wvT, QH, KH, VT);
  attn_kernel<<<BB * NH * (SS / 16), 128, 0, stream>>>(QH, KH, VT, mask, attn_out,
                                                       concat);
  outproj_kernel<<<BB * (SS / 16) * 4, 128, 0, stream>>>(concat, pwbf, proj_b, q, z);
  ln_kernel<<<BB * SS, 256, 0, stream>>>(z, gamma, beta, ln_out);
}
